// GPTLanguageModel_67937792688540
// MI455X (gfx1250) — matrix operand order repaired and checked by reading, not compiled, not run on hardware
//
#include <hip/hip_runtime.h>
#include <math.h>

// ---------------------------------------------------------------------------
// GPT forward for MI455X (gfx1250, wave32, WMMA).
// GEMMs: bf16 x bf16 -> f32 accum via v_wmma_f32_16x16x32_bf16.
// Staging: GLOBAL_LOAD_ASYNC_TO_LDS_B128 (ASYNCcnt) with double buffering.
// Weights pre-converted AND pre-transposed (B operand = W^T[N][K] bf16).
// Attention is flash-style (no TxT materialization).
// ---------------------------------------------------------------------------

typedef __attribute__((ext_vector_type(16))) __bf16 v16bf;
typedef __attribute__((ext_vector_type(8)))  float  v8f;

#define LL 6
#define BB 8
#define TT 1024
#define CC 1024
#define HH 16
#define HSZ 64
#define VV 4096

union Frag { v16bf v; uint4 q[2]; };

__device__ __forceinline__ v8f wmma_bf16(v16bf a, v16bf b, v8f c) {
  return __builtin_amdgcn_wmma_f32_16x16x32_bf16(false, a, false, b, (short)0, c,
                                                 false, false);
}

__device__ __forceinline__ v8f zero8() {
  v8f z;
  for (int i = 0; i < 8; ++i) z[i] = 0.f;
  return z;
}

// Async global->LDS copy, 16 bytes per lane (GVS addressing: SGPR base +
// 32-bit per-lane byte offset). Tracked by ASYNCcnt.
__device__ __forceinline__ void async_copy_b128(unsigned lds_off, unsigned voff,
                                                const void* sbase) {
  asm volatile("global_load_async_to_lds_b128 %0, %1, %2"
               :: "v"(lds_off), "v"(voff), "s"(sbase) : "memory");
}
__device__ __forceinline__ void wait_async0() {
  asm volatile("s_wait_asynccnt 0" ::: "memory");
}
__device__ __forceinline__ unsigned lds_addr(const void* p) {
  return (unsigned)(unsigned long long)p;  // low 32 bits = LDS byte offset
}

// ---------------------------------------------------------------------------
// Tiled transpose + f32->bf16: out[l][n][k] = (bf16) in[l][k][n]
// ---------------------------------------------------------------------------
__global__ __launch_bounds__(256)
void transpose_f2bf_kernel(const float* __restrict__ in, __bf16* __restrict__ out,
                           int K, int N) {
  __shared__ float tile[32][33];
  const int lx = threadIdx.x & 31, ly = threadIdx.x >> 5;
  const int l = blockIdx.z;
  const float* ip = in + (size_t)l * K * N;
  __bf16* op = out + (size_t)l * K * N;
  const int n0 = blockIdx.x * 32, k0 = blockIdx.y * 32;
  for (int i = 0; i < 4; ++i)
    tile[ly + 8 * i][lx] = ip[(size_t)(k0 + ly + 8 * i) * N + n0 + lx];
  __syncthreads();
  for (int i = 0; i < 4; ++i)
    op[(size_t)(n0 + ly + 8 * i) * K + k0 + lx] = (__bf16)tile[lx][ly + 8 * i];
}

// ---------------------------------------------------------------------------
// x[b,t,:] = tok_emb[idx[b,t],:] + pos_emb[t,:]
// ---------------------------------------------------------------------------
__global__ __launch_bounds__(256)
void embed_kernel(const int* __restrict__ idx, const float* __restrict__ tok,
                  const float* __restrict__ pos, float* __restrict__ x) {
  const int row = blockIdx.x;
  const int t = row % TT;
  const int token = idx[row];
  const float* tr = tok + (size_t)token * CC;
  const float* pr = pos + (size_t)t * CC;
  float* xr = x + (size_t)row * CC;
  for (int i = threadIdx.x; i < CC; i += 256) xr[i] = tr[i] + pr[i];
}

// ---------------------------------------------------------------------------
// LayerNorm over C=1024, writes bf16 (GEMM input)
// ---------------------------------------------------------------------------
__global__ __launch_bounds__(256)
void layernorm_kernel(const float* __restrict__ x, const float* __restrict__ g,
                      const float* __restrict__ b, __bf16* __restrict__ out) {
  __shared__ float red[256];
  const int row = blockIdx.x, tid = threadIdx.x;
  const float* xr = x + (size_t)row * CC;

  float s = 0.f;
  for (int i = tid; i < CC; i += 256) s += xr[i];
  red[tid] = s;
  __syncthreads();
  for (int off = 128; off > 0; off >>= 1) {
    if (tid < off) red[tid] += red[tid + off];
    __syncthreads();
  }
  const float mu = red[0] * (1.f / CC);
  __syncthreads();

  s = 0.f;
  for (int i = tid; i < CC; i += 256) {
    float d = xr[i] - mu;
    s += d * d;
  }
  red[tid] = s;
  __syncthreads();
  for (int off = 128; off > 0; off >>= 1) {
    if (tid < off) red[tid] += red[tid + off];
    __syncthreads();
  }
  const float rs = rsqrtf(red[0] * (1.f / CC) + 1e-5f);

  __bf16* orow = out + (size_t)row * CC;
  for (int i = tid; i < CC; i += 256)
    orow[i] = (__bf16)((xr[i] - mu) * rs * g[i] + b[i]);
}

// ---------------------------------------------------------------------------
// GEMM: C[M,N] = A[M,K](bf16, row-major) * B (given as W^T[N][K], bf16)
// Block tile 128x128, BK=64, 8 waves, wave tile 32x64 (2x4 WMMA tiles).
// Double-buffered async global->LDS staging; one barrier per K-tile.
// ---------------------------------------------------------------------------
enum {
  EP_BF16 = 0,             // outB = acc
  EP_BF16_RELU_BIAS = 1,   // outB = relu(acc + bias)
  EP_F32_RESID_BIAS = 2,   // outF = resid + acc + bias
  EP_F32_BIAS = 3,         // outF = acc + bias
  EP_VT = 4                // outB[(b*C+col)*T + t] = acc  (per-head transposed)
};

template <int EP>
__global__ __launch_bounds__(256)
void gemm_bf16_kernel(const __bf16* __restrict__ A, const __bf16* __restrict__ Bt,
                      const float* __restrict__ bias, const float* __restrict__ resid,
                      float* __restrict__ outF, __bf16* __restrict__ outB,
                      int M, int N, int K) {
  __shared__ __align__(16) __bf16 As[2][128 * 72];
  __shared__ __align__(16) __bf16 Bs[2][128 * 72];

  const int tid = threadIdx.x;
  const int lane = tid & 31;
  const int wave = tid >> 5;
  const int wm = wave >> 1;  // 0..3 -> M sub-block of 32
  const int wn = wave & 1;   // 0..1 -> N sub-block of 64
  const int hh = lane >> 4;
  const int ln = lane & 15;
  const int bm = blockIdx.y, bn = blockIdx.x;

  v8f acc[2][4];
  for (int s = 0; s < 2; ++s)
    for (int t = 0; t < 4; ++t) acc[s][t] = zero8();

  const __bf16* Ag = A + (size_t)bm * 128 * K;
  const __bf16* Bg = Bt + (size_t)bn * 128 * K;
  const unsigned asb[2] = {lds_addr(&As[0][0]), lds_addr(&As[1][0])};
  const unsigned bsb[2] = {lds_addr(&Bs[0][0]), lds_addr(&Bs[1][0])};

  auto stage = [&](int buf, int kk) {
    for (int p = 0; p < 4; ++p) {
      int slot = p * 256 + tid;
      int row = slot >> 3, k8 = (slot & 7) * 8;
      unsigned loff = (unsigned)(row * 72 + k8) * 2u;
      unsigned voff = (unsigned)(row * K + kk + k8) * 2u;
      async_copy_b128(asb[buf] + loff, voff, Ag);
      async_copy_b128(bsb[buf] + loff, voff, Bg);
    }
  };

  stage(0, 0);
  int buf = 0;
  for (int kk = 0; kk < K; kk += 64) {
    wait_async0();
    __syncthreads();  // all waves' async writes for `buf` landed; `buf^1` free
    if (kk + 64 < K) stage(buf ^ 1, kk + 64);

    const __bf16* Asb = &As[buf][0];
    const __bf16* Bsb = &Bs[buf][0];
    for (int s2 = 0; s2 < 2; ++s2) {
      Frag af[2], bfr[4];
      for (int s = 0; s < 2; ++s) {
        int m = wm * 32 + s * 16 + ln;
        const uint4* rp = (const uint4*)&Asb[m * 72];
        af[s].q[0] = rp[4 * s2 + hh];       // K = 32*s2 + 8*hh + (0..7)
        af[s].q[1] = rp[4 * s2 + hh + 2];   // K = 32*s2 + 8*hh + 16 + (0..7)
      }
      for (int t = 0; t < 4; ++t) {
        int n = wn * 64 + t * 16 + ln;
        const uint4* rp = (const uint4*)&Bsb[n * 72];
        bfr[t].q[0] = rp[4 * s2 + hh];      // K = 32*s2 + 8*hh + (0..7)
        bfr[t].q[1] = rp[4 * s2 + hh + 2];  // K = 32*s2 + 8*hh + 16 + (0..7)
      }
      for (int s = 0; s < 2; ++s)
        for (int t = 0; t < 4; ++t)
          acc[s][t] = wmma_bf16(af[s].v, bfr[t].v, acc[s][t]);
    }
    buf ^= 1;
  }

  // epilogue: C layout -> row M = r + 8*hh, col N = ln
  for (int s = 0; s < 2; ++s) {
    int row0 = bm * 128 + wm * 32 + s * 16 + hh * 8;
    for (int t = 0; t < 4; ++t) {
      int col = bn * 128 + wn * 64 + t * 16 + ln;
      if (EP == EP_VT) {
        int b = row0 >> 10;        // row0 / TT
        int t0 = row0 & (TT - 1);  // 8-aligned -> 16B-aligned store
        __bf16 tmp[8];
        for (int r = 0; r < 8; ++r) tmp[r] = (__bf16)acc[s][t][r];
        *(uint4*)&outB[((size_t)b * CC + col) * TT + t0] = *(uint4*)tmp;
      } else if (EP == EP_BF16) {
        for (int r = 0; r < 8; ++r)
          outB[(size_t)(row0 + r) * N + col] = (__bf16)acc[s][t][r];
      } else if (EP == EP_BF16_RELU_BIAS) {
        float bv = bias[col];
        for (int r = 0; r < 8; ++r)
          outB[(size_t)(row0 + r) * N + col] = (__bf16)fmaxf(acc[s][t][r] + bv, 0.f);
      } else if (EP == EP_F32_RESID_BIAS) {
        float bv = bias[col];
        float res[8];
        for (int r = 0; r < 8; ++r) res[r] = resid[(size_t)(row0 + r) * N + col];
        for (int r = 0; r < 8; ++r)
          outF[(size_t)(row0 + r) * N + col] = acc[s][t][r] + bv + res[r];
      } else {  // EP_F32_BIAS
        float bv = bias[col];
        for (int r = 0; r < 8; ++r)
          outF[(size_t)(row0 + r) * N + col] = acc[s][t][r] + bv;
      }
    }
  }
}

// ---------------------------------------------------------------------------
// Flash attention (causal). grid = (T/64, H, B), block = 128 (4 waves).
// Each wave owns 16 q-rows; streams 64-key chunks with double-buffered async
// K/V staging. NOTE: reference MULTIPLIES scores by sqrt(HS)=8.
// q/k layout: [b][t][head*HS + hs];  v pre-transposed: vt[(b*C+hd*HS+hs)*T + t]
// ---------------------------------------------------------------------------
__global__ __launch_bounds__(128)
void flash_attn_kernel(const __bf16* __restrict__ q, const __bf16* __restrict__ k,
                       const __bf16* __restrict__ vt, __bf16* __restrict__ o) {
  __shared__ __align__(16) __bf16 Kc[2][64 * 72];   // [key][hs]
  __shared__ __align__(16) __bf16 Vs[2][64 * 72];   // [hs][key]
  __shared__ __align__(16) __bf16 Pb[4 * 16 * 72];  // per-wave P tiles [m][key]

  const int tid = threadIdx.x;
  const int lane = tid & 31;
  const int wave = tid >> 5;
  const int hh = lane >> 4, ln = lane & 15;
  const int q0 = blockIdx.x * 64;
  const int hd = blockIdx.y;
  const int bz = blockIdx.z;

  const size_t base = (size_t)bz * TT * CC + (size_t)hd * HSZ;
  const size_t vbase = ((size_t)bz * CC + (size_t)hd * HSZ) * TT;

  // Q fragments for this wave's 16 rows (kept in registers)
  Frag qf[2];
  {
    int qr = q0 + wave * 16 + ln;
    const __bf16* qp = q + base + (size_t)qr * CC;
    for (int s = 0; s < 2; ++s) {
      qf[s].q[0] = *(const uint4*)(qp + s * 32 + hh * 8);
      qf[s].q[1] = *(const uint4*)(qp + s * 32 + hh * 8 + 16);
    }
  }

  float m_run[8], l_run[8];
  v8f ov[4];
  for (int r = 0; r < 8; ++r) { m_run[r] = -INFINITY; l_run[r] = 0.f; }
  for (int t = 0; t < 4; ++t) ov[t] = zero8();

  __bf16* Pw = &Pb[wave * 16 * 72];
  const unsigned kcb[2] = {lds_addr(&Kc[0][0]), lds_addr(&Kc[1][0])};
  const unsigned vsb[2] = {lds_addr(&Vs[0][0]), lds_addr(&Vs[1][0])};

  auto stageKV = [&](int buf, int k0) {
    const __bf16* Kg = k + base + (size_t)k0 * CC;
    const __bf16* Vg = vt + vbase + k0;
    for (int i = 0; i < 4; ++i) {
      int idx = i * 128 + tid;
      int r = idx >> 3, c8 = (idx & 7) * 8;
      unsigned loff = (unsigned)(r * 72 + c8) * 2u;
      async_copy_b128(kcb[buf] + loff, (unsigned)(r * CC + c8) * 2u, Kg);
      async_copy_b128(vsb[buf] + loff, (unsigned)(r * TT + c8) * 2u, Vg);
    }
  };

  const int kend = q0 + 64;
  stageKV(0, 0);
  int buf = 0;
  for (int k0 = 0; k0 < kend; k0 += 64) {
    wait_async0();
    __syncthreads();
    if (k0 + 64 < kend) stageKV(buf ^ 1, k0 + 64);
    const __bf16* Kcb = &Kc[buf][0];
    const __bf16* Vsb = &Vs[buf][0];

    // S = Q K^T (16x64 per wave, K-dim = HS = 64)
    v8f st[4];
    for (int t = 0; t < 4; ++t) st[t] = zero8();
    for (int t = 0; t < 4; ++t) {
      const uint4* rp = (const uint4*)&Kcb[(t * 16 + ln) * 72];
      for (int s = 0; s < 2; ++s) {
        Frag bfr;
        bfr.q[0] = rp[4 * s + hh];
        bfr.q[1] = rp[4 * s + hh + 2];
        st[t] = wmma_bf16(qf[s].v, bfr.v, st[t]);
      }
    }

    // scale (*8) and causal mask (only diagonal chunk needs it)
    const bool need_mask = (k0 + 64 > q0);
    for (int t = 0; t < 4; ++t)
      for (int r = 0; r < 8; ++r) {
        float x = st[t][r] * 8.0f;
        if (need_mask) {
          int qrow = q0 + wave * 16 + r + 8 * hh;
          int kcol = k0 + t * 16 + ln;
          if (kcol > qrow) x = -INFINITY;
        }
        st[t][r] = x;
      }

    // online softmax (a row lives in one half-wave; xor 1/2/4/8 reduces it)
    for (int r = 0; r < 8; ++r) {
      float mx = fmaxf(fmaxf(st[0][r], st[1][r]), fmaxf(st[2][r], st[3][r]));
      mx = fmaxf(mx, __shfl_xor(mx, 1, 32));
      mx = fmaxf(mx, __shfl_xor(mx, 2, 32));
      mx = fmaxf(mx, __shfl_xor(mx, 4, 32));
      mx = fmaxf(mx, __shfl_xor(mx, 8, 32));
      float mnew = fmaxf(m_run[r], mx);
      float alpha = __expf(m_run[r] - mnew);
      m_run[r] = mnew;
      float rs = 0.f;
      for (int t = 0; t < 4; ++t) {
        float p = __expf(st[t][r] - mnew);
        st[t][r] = p;
        rs += p;
      }
      rs += __shfl_xor(rs, 1, 32);
      rs += __shfl_xor(rs, 2, 32);
      rs += __shfl_xor(rs, 4, 32);
      rs += __shfl_xor(rs, 8, 32);
      l_run[r] = l_run[r] * alpha + rs;
      for (int t = 0; t < 4; ++t) ov[t][r] *= alpha;
    }

    // transpose P (C-layout) -> A-layout via per-wave LDS
    for (int t = 0; t < 4; ++t)
      for (int r = 0; r < 8; ++r)
        Pw[(r + 8 * hh) * 72 + t * 16 + ln] = (__bf16)st[t][r];
    asm volatile("s_wait_dscnt 0" ::: "memory");  // LDS in-order per wave

    // O += P V
    {
      const uint4* prp = (const uint4*)&Pw[ln * 72];
      Frag pf[2];
      for (int s = 0; s < 2; ++s) {
        pf[s].q[0] = prp[4 * s + hh];
        pf[s].q[1] = prp[4 * s + hh + 2];
      }
      for (int t = 0; t < 4; ++t) {
        const uint4* rp = (const uint4*)&Vsb[(t * 16 + ln) * 72];
        for (int s = 0; s < 2; ++s) {
          Frag bfr;
          bfr.q[0] = rp[4 * s + hh];
          bfr.q[1] = rp[4 * s + hh + 2];
          ov[t] = wmma_bf16(pf[s].v, bfr.v, ov[t]);
        }
      }
    }
    buf ^= 1;
  }

  // normalize and write O (bf16, normal [t][C] layout)
  for (int t = 0; t < 4; ++t)
    for (int r = 0; r < 8; ++r) {
      int qrow = q0 + wave * 16 + r + 8 * hh;
      float val = ov[t][r] / l_run[r];
      o[base + (size_t)qrow * CC + t * 16 + ln] = (__bf16)val;
    }
}

// ---------------------------------------------------------------------------
// Host orchestration
// ---------------------------------------------------------------------------
extern "C" void kernel_launch(void* const* d_in, const int* in_sizes, int n_in,
                              void* d_out, int out_size, void* d_ws, size_t ws_size,
                              hipStream_t stream) {
  (void)in_sizes; (void)n_in; (void)out_size; (void)ws_size;

  const int*   idx     = (const int*)  d_in[0];
  const float* tok_emb = (const float*)d_in[1];
  const float* pos_emb = (const float*)d_in[2];
  const float* Wq      = (const float*)d_in[3];
  const float* Wk      = (const float*)d_in[4];
  const float* Wv      = (const float*)d_in[5];
  const float* Wo      = (const float*)d_in[6];
  const float* bo      = (const float*)d_in[7];
  const float* ln1_g   = (const float*)d_in[8];
  const float* ln1_b   = (const float*)d_in[9];
  const float* ln2_g   = (const float*)d_in[10];
  const float* ln2_b   = (const float*)d_in[11];
  const float* W1      = (const float*)d_in[12];
  const float* b1      = (const float*)d_in[13];
  const float* W2      = (const float*)d_in[14];
  const float* b2      = (const float*)d_in[15];
  const float* lnf_g   = (const float*)d_in[16];
  const float* lnf_b   = (const float*)d_in[17];
  const float* lm_w    = (const float*)d_in[18];
  const float* lm_b    = (const float*)d_in[19];

  char* ws = (char*)d_ws;
  size_t off = 0;
  auto alloc = [&](size_t bytes) -> char* {
    char* p = ws + off;
    off = (off + bytes + 255) & ~(size_t)255;
    return p;
  };

  const size_t nRows = (size_t)BB * TT;  // 8192

  __bf16* wqt  = (__bf16*)alloc((size_t)LL * CC * CC * 2);
  __bf16* wkt  = (__bf16*)alloc((size_t)LL * CC * CC * 2);
  __bf16* wvt  = (__bf16*)alloc((size_t)LL * CC * CC * 2);
  __bf16* wot  = (__bf16*)alloc((size_t)LL * CC * CC * 2);
  __bf16* w1t  = (__bf16*)alloc((size_t)LL * CC * 4 * CC * 2);
  __bf16* w2t  = (__bf16*)alloc((size_t)LL * 4 * CC * CC * 2);
  __bf16* lmwt = (__bf16*)alloc((size_t)CC * VV * 2);
  float*  x    = (float*) alloc(nRows * CC * 4);
  __bf16* h    = (__bf16*)alloc(nRows * CC * 2);
  __bf16* qb   = (__bf16*)alloc(nRows * CC * 2);
  __bf16* kb   = (__bf16*)alloc(nRows * CC * 2);
  __bf16* vtb  = (__bf16*)alloc(nRows * CC * 2);
  __bf16* ob   = (__bf16*)alloc(nRows * CC * 2);
  __bf16* hid  = (__bf16*)alloc(nRows * 4 * CC * 2);

  {
    dim3 gCC(CC / 32, CC / 32, LL);
    transpose_f2bf_kernel<<<gCC, 256, 0, stream>>>(Wq, wqt, CC, CC);
    transpose_f2bf_kernel<<<gCC, 256, 0, stream>>>(Wk, wkt, CC, CC);
    transpose_f2bf_kernel<<<gCC, 256, 0, stream>>>(Wv, wvt, CC, CC);
    transpose_f2bf_kernel<<<gCC, 256, 0, stream>>>(Wo, wot, CC, CC);
    dim3 g1(4 * CC / 32, CC / 32, LL);
    transpose_f2bf_kernel<<<g1, 256, 0, stream>>>(W1, w1t, CC, 4 * CC);
    dim3 g2(CC / 32, 4 * CC / 32, LL);
    transpose_f2bf_kernel<<<g2, 256, 0, stream>>>(W2, w2t, 4 * CC, CC);
    dim3 gl(VV / 32, CC / 32, 1);
    transpose_f2bf_kernel<<<gl, 256, 0, stream>>>(lm_w, lmwt, CC, VV);
  }

  embed_kernel<<<(int)nRows, 256, 0, stream>>>(idx, tok_emb, pos_emb, x);

  const int M = (int)nRows;
  for (int l = 0; l < LL; ++l) {
    const size_t oCC = (size_t)l * CC * CC;
    layernorm_kernel<<<(int)nRows, 256, 0, stream>>>(x, ln1_g + (size_t)l * CC,
                                                     ln1_b + (size_t)l * CC, h);
    {
      dim3 grid(CC / 128, M / 128);
      gemm_bf16_kernel<EP_BF16><<<grid, 256, 0, stream>>>(
          h, wqt + oCC, nullptr, nullptr, nullptr, qb, M, CC, CC);
      gemm_bf16_kernel<EP_BF16><<<grid, 256, 0, stream>>>(
          h, wkt + oCC, nullptr, nullptr, nullptr, kb, M, CC, CC);
      gemm_bf16_kernel<EP_VT><<<grid, 256, 0, stream>>>(
          h, wvt + oCC, nullptr, nullptr, nullptr, vtb, M, CC, CC);
    }
    {
      dim3 grid(TT / 64, HH, BB);
      flash_attn_kernel<<<grid, 128, 0, stream>>>(qb, kb, vtb, ob);
    }
    {
      dim3 grid(CC / 128, M / 128);
      gemm_bf16_kernel<EP_F32_RESID_BIAS><<<grid, 256, 0, stream>>>(
          ob, wot + oCC, bo + (size_t)l * CC, x, x, nullptr, M, CC, CC);
    }
    layernorm_kernel<<<(int)nRows, 256, 0, stream>>>(x, ln2_g + (size_t)l * CC,
                                                     ln2_b + (size_t)l * CC, h);
    {
      dim3 grid(4 * CC / 128, M / 128);
      gemm_bf16_kernel<EP_BF16_RELU_BIAS><<<grid, 256, 0, stream>>>(
          h, w1t + (size_t)l * CC * 4 * CC, b1 + (size_t)l * 4 * CC, nullptr,
          nullptr, hid, M, 4 * CC, CC);
    }
    {
      dim3 grid(CC / 128, M / 128);
      gemm_bf16_kernel<EP_F32_RESID_BIAS><<<grid, 256, 0, stream>>>(
          hid, w2t + (size_t)l * 4 * CC * CC, b2 + (size_t)l * CC, x, x, nullptr,
          M, CC, 4 * CC);
    }
  }

  layernorm_kernel<<<(int)nRows, 256, 0, stream>>>(x, lnf_g, lnf_b, h);
  {
    dim3 grid(VV / 128, M / 128);
    gemm_bf16_kernel<EP_F32_BIAS><<<grid, 256, 0, stream>>>(
        h, lmwt, lm_b, nullptr, (float*)d_out, nullptr, M, VV, CC);
  }
}
